// MultiHeadedAttention_75419625718115
// MI455X (gfx1250) — compile-verified, matrix operand order checked by reading, not run on hardware
//
#include <hip/hip_runtime.h>
#include <hip/hip_bf16.h>
#include <math.h>

// ---------------------------------------------------------------------------
// MultiHeadedAttention for MI455X (gfx1250, wave32, WMMA)
//   B=2, S=2048, D=1024, H=16, DK=64
//   q/k/v projections: f16 WMMA 16x16x32, f32 accumulate, double-buffered LDS
//   attention: flash-style online softmax, WMMA for QK^T and PV,
//              double-buffered K/V staging, packed-b32 P/V LDS traffic
//   final: residual + LayerNorm (f32)
// ---------------------------------------------------------------------------

typedef __attribute__((ext_vector_type(16))) _Float16 v16h;
typedef __attribute__((ext_vector_type(8)))  _Float16 v8h;
typedef __attribute__((ext_vector_type(4)))  _Float16 v4h;
typedef __attribute__((ext_vector_type(8)))  float    v8f;

static __device__ __forceinline__ v16h cat8(v8h lo, v8h hi) {
    return __builtin_shufflevector(lo, hi, 0,1,2,3,4,5,6,7,8,9,10,11,12,13,14,15);
}

static __device__ __forceinline__ v8f wmma16(v16h a, v16h b, v8f c) {
    // D = A(16x32 f16) * B(32x16 f16) + C(16x16 f32)
    return __builtin_amdgcn_wmma_f32_16x16x32_f16(
        /*neg_a=*/false, a, /*neg_b=*/false, b,
        /*c_mod=*/(short)0, c, /*reuse_a=*/false, /*reuse_b=*/false);
}

static __device__ __forceinline__ unsigned pack_f2h2(float a, float b) {
    union { _Float16 h[2]; unsigned u; } p;
    p.h[0] = (_Float16)a; p.h[1] = (_Float16)b;
    return p.u;
}

constexpr int Bc  = 2;
constexpr int Sc  = 2048;
constexpr int Dc  = 1024;
constexpr int Hc  = 16;
constexpr int DKc = 64;
constexpr int Mtot = Bc * Sc;   // 4096

// ---------------------------------------------------------------------------
// Projection GEMM: Out[b,h,s,dk] = (X[m,:] @ W[:,n] + bias[n]) as f16
//   Block tile 128x128, BK=32, 256 threads = 8 waves (4 x 2 wave grid),
//   each wave: 32x64 region = 2x4 tiles of 16x16 WMMA. Double-buffered LDS.
// ---------------------------------------------------------------------------
__global__ __launch_bounds__(256)
void proj_gemm_kernel(const float* __restrict__ X, const float* __restrict__ W,
                      const float* __restrict__ bias, _Float16* __restrict__ Out) {
    __shared__ __align__(16) _Float16 As[2][128][40];  // [buf][m][k]
    __shared__ __align__(16) _Float16 Bs[2][128][40];  // [buf][n][k] (W transposed)

    const int tid   = threadIdx.x;
    const int lane  = tid & 31;
    const int wv    = tid >> 5;       // 0..7
    const int waveM = wv >> 1;        // 0..3
    const int waveN = wv & 1;         // 0..1
    const int l16   = lane & 15;
    const int hi    = lane >> 4;      // 0/1
    const int m0    = blockIdx.y * 128;
    const int n0    = blockIdx.x * 128;

    v8f acc[2][4];
#pragma unroll
    for (int i = 0; i < 2; ++i)
#pragma unroll
        for (int j = 0; j < 4; ++j) acc[i][j] = (v8f)0.0f;

    const int arow = tid >> 1;             // 0..127
    const int acol = (tid & 1) * 16;       // 0 or 16

    float4 ra[4];        // A prefetch regs (16 floats)
    float4 rb[4];        // B prefetch regs (16 floats)

    auto load_tile = [&](int k0) {
        const float4* xp = (const float4*)(X + (size_t)(m0 + arow) * Dc + k0 + acol);
        ra[0] = xp[0]; ra[1] = xp[1]; ra[2] = xp[2]; ra[3] = xp[3];
#pragma unroll
        for (int j = 0; j < 4; ++j) {
            int u  = tid + j * 256;        // 0..1023 float4 units
            int kr = u >> 5;               // 0..31
            int c4 = u & 31;               // 0..31
            rb[j] = *(const float4*)(W + (size_t)(k0 + kr) * Dc + n0 + c4 * 4);
        }
    };
    auto store_tile = [&](int buf) {
        float fb[16] = {ra[0].x,ra[0].y,ra[0].z,ra[0].w, ra[1].x,ra[1].y,ra[1].z,ra[1].w,
                        ra[2].x,ra[2].y,ra[2].z,ra[2].w, ra[3].x,ra[3].y,ra[3].z,ra[3].w};
        v8h h0, h1;
#pragma unroll
        for (int j = 0; j < 8; ++j) { h0[j] = (_Float16)fb[j]; h1[j] = (_Float16)fb[8+j]; }
        *(v8h*)&As[buf][arow][acol]     = h0;
        *(v8h*)&As[buf][arow][acol + 8] = h1;
#pragma unroll
        for (int j = 0; j < 4; ++j) {
            int u  = tid + j * 256;
            int kr = u >> 5;
            int c4 = u & 31;
            Bs[buf][c4*4 + 0][kr] = (_Float16)rb[j].x;
            Bs[buf][c4*4 + 1][kr] = (_Float16)rb[j].y;
            Bs[buf][c4*4 + 2][kr] = (_Float16)rb[j].z;
            Bs[buf][c4*4 + 3][kr] = (_Float16)rb[j].w;
        }
    };

    constexpr int NKT = Dc / 32;    // 32 k-tiles
    load_tile(0);
    store_tile(0);
    __syncthreads();

    for (int kt = 0; kt < NKT; ++kt) {
        const int cur = kt & 1;
        if (kt + 1 < NKT) load_tile((kt + 1) * 32);   // global loads overlap WMMAs

        v16h afrag[2];
#pragma unroll
        for (int tm = 0; tm < 2; ++tm) {
            int row = waveM*32 + tm*16 + l16;
            v8h lo = *(const v8h*)&As[cur][row][hi*8];
            v8h hh = *(const v8h*)&As[cur][row][hi*8 + 16];
            afrag[tm] = cat8(lo, hh);
        }
#pragma unroll
        for (int tn = 0; tn < 4; ++tn) {
            int n = waveN*64 + tn*16 + l16;
            v8h lo = *(const v8h*)&Bs[cur][n][hi*8];
            v8h hh = *(const v8h*)&Bs[cur][n][hi*8 + 16];
            v16h bfrag = cat8(lo, hh);
#pragma unroll
            for (int tm = 0; tm < 2; ++tm)
                acc[tm][tn] = wmma16(afrag[tm], bfrag, acc[tm][tn]);
        }

        if (kt + 1 < NKT) store_tile(cur ^ 1);
        __syncthreads();
    }

    // ---- epilogue: bias add, scatter to [B,H,S,DK] f16 ----
#pragma unroll
    for (int tn = 0; tn < 4; ++tn) {
        int gn = n0 + waveN*64 + tn*16 + l16;
        float bv = bias[gn];
        int h = gn >> 6;       // head
        int d = gn & 63;       // within head
#pragma unroll
        for (int tm = 0; tm < 2; ++tm) {
#pragma unroll
            for (int r = 0; r < 8; ++r) {
                int gm   = m0 + waveM*32 + tm*16 + r + hi*8;
                int bidx = gm >> 11;          // S = 2048
                int sidx = gm & 2047;
                size_t off = ((size_t)((bidx*Hc + h)) * Sc + sidx) * DKc + d;
                Out[off] = (_Float16)(acc[tm][tn][r] + bv);
            }
        }
    }
}

// ---------------------------------------------------------------------------
// Flash attention: one block per (b, h, 128 q-rows); 8 waves x 16 q-rows.
// K loop: 64 chunks of 32 keys, double-buffered. Online softmax.
// The PV contraction index sk is stored PERMUTED as pos = 2*(sk%16)+(sk/16)
// in both Ps and Vs (consistently), enabling packed b32 LDS stores.
// ---------------------------------------------------------------------------
__global__ __launch_bounds__(256)
void attention_kernel(const _Float16* __restrict__ qh, const _Float16* __restrict__ kh,
                      const _Float16* __restrict__ vh, float* __restrict__ attn_out) {
    __shared__ __align__(16) _Float16 Ks[2][32][72];     // [buf][sk][dk]
    __shared__ __align__(16) _Float16 Vs[2][64][40];     // [buf][dv][sk-permuted]
    __shared__ __align__(16) _Float16 Ps[8][16][40];     // per-wave P bounce [m][sk-permuted]

    const int tid  = threadIdx.x;
    const int lane = tid & 31;
    const int wv   = tid >> 5;
    const int l16  = lane & 15;
    const int hi   = lane >> 4;

    const int b  = blockIdx.z;
    const int h  = blockIdx.y;
    const int q0 = blockIdx.x * 128;

    const size_t headoff = (size_t)(b*Hc + h) * Sc * DKc;
    const _Float16* Q  = qh + headoff;
    const _Float16* Kp = kh + headoff;
    const _Float16* Vp = vh + headoff;

    // Q fragments (prescaled by 1/sqrt(DK) = 0.125), resident for whole kernel
    v16h qf[2];
    {
        int qrow = q0 + wv*16 + l16;
#pragma unroll
        for (int c = 0; c < 2; ++c) {
            const _Float16* qp = Q + (size_t)qrow * DKc + c*32 + hi*8;
            v8h lo = *(const v8h*)qp;
            v8h hh = *(const v8h*)(qp + 16);
            v16h f = cat8(lo, hh);
#pragma unroll
            for (int j = 0; j < 16; ++j) f[j] = (_Float16)((float)f[j] * 0.125f);
            qf[c] = f;
        }
    }

    float mrow[8], lrow[8];
#pragma unroll
    for (int r = 0; r < 8; ++r) { mrow[r] = -1.0e30f; lrow[r] = 0.0f; }
    v8f o[4];
#pragma unroll
    for (int t = 0; t < 4; ++t) o[t] = (v8f)0.0f;

    // staging thread mappings
    const int ksk  = tid >> 3;        // 0..31 : K row
    const int kc8  = (tid & 7) * 8;   // 0..56 : K col
    const int vskp = tid >> 4;        // 0..15 : V sk-pair (sk, sk+16)
    const int vg   = tid & 15;        // 0..15 : V dv group of 4

    v8h kreg; v4h vr0, vr1;           // prefetch regs
    auto load_kv = [&](int sk0) {
        kreg = *(const v8h*)(Kp + (size_t)(sk0 + ksk) * DKc + kc8);
        vr0  = *(const v4h*)(Vp + (size_t)(sk0 + vskp)      * DKc + vg*4);
        vr1  = *(const v4h*)(Vp + (size_t)(sk0 + vskp + 16) * DKc + vg*4);
    };
    auto store_kv = [&](int buf) {
        *(v8h*)&Ks[buf][ksk][kc8] = kreg;
#pragma unroll
        for (int i = 0; i < 4; ++i) {
            union { _Float16 h[2]; unsigned u; } p;
            p.h[0] = vr0[i]; p.h[1] = vr1[i];          // sk and sk+16 -> adjacent permuted cols
            *(unsigned*)&Vs[buf][vg*4 + i][2*vskp] = p.u;
        }
    };

    constexpr int NCH = Sc / 32;      // 64 chunks
    load_kv(0);
    store_kv(0);
    __syncthreads();

    for (int c = 0; c < NCH; ++c) {
        const int cur = c & 1;
        if (c + 1 < NCH) load_kv((c + 1) * 32);   // overlap next chunk with compute

        // ---- scores: 16 q-rows x 32 keys = 2 C-tiles, K=64 -> 4 WMMAs ----
        v8f sc[2] = { (v8f)0.0f, (v8f)0.0f };
#pragma unroll
        for (int tc = 0; tc < 2; ++tc) {
            int n = tc*16 + l16;   // key column (natural sk order)
#pragma unroll
            for (int kc = 0; kc < 2; ++kc) {
                const _Float16* kp2 = &Ks[cur][n][kc*32 + hi*8];
                v8h lo = *(const v8h*)kp2;
                v8h hh = *(const v8h*)(kp2 + 16);
                sc[tc] = wmma16(qf[kc], cat8(lo, hh), sc[tc]);
            }
        }

        // ---- online softmax (row = r + 8*hi, reductions within half-wave) ----
#pragma unroll
        for (int r = 0; r < 8; ++r) {
            float a0 = sc[0][r], a1 = sc[1][r];
            float t = fmaxf(a0, a1);
#pragma unroll
            for (int off = 1; off < 16; off <<= 1) t = fmaxf(t, __shfl_xor(t, off));
            float mnew = fmaxf(mrow[r], t);
            float fscale = __expf(mrow[r] - mnew);
            float e0 = __expf(a0 - mnew);     // sk = l16       -> permuted col 2*l16
            float e1 = __expf(a1 - mnew);     // sk = 16 + l16  -> permuted col 2*l16+1
            float rs = e0 + e1;
#pragma unroll
            for (int off = 1; off < 16; off <<= 1) rs += __shfl_xor(rs, off);
            lrow[r] = lrow[r] * fscale + rs;
            mrow[r] = mnew;
#pragma unroll
            for (int tn = 0; tn < 4; ++tn) o[tn][r] *= fscale;
            int m = r + hi*8;
            *(unsigned*)&Ps[wv][m][2*l16] = pack_f2h2(e0, e1);   // one b32 store
        }
        // (no barrier: Ps region is wave-private, DS ops in-order within a wave)

        // ---- PV: O(16x64) += P(16x32) @ V(32x64) -> 4 WMMAs ----
        {
            const _Float16* pp = &Ps[wv][l16][hi*8];
            v8h plo = *(const v8h*)pp;
            v8h phi = *(const v8h*)(pp + 16);
            v16h pf = cat8(plo, phi);
#pragma unroll
            for (int tn = 0; tn < 4; ++tn) {
                const _Float16* vp2 = &Vs[cur][tn*16 + l16][hi*8];
                v8h vlo = *(const v8h*)vp2;
                v8h vhi = *(const v8h*)(vp2 + 16);
                o[tn] = wmma16(pf, cat8(vlo, vhi), o[tn]);
            }
        }

        if (c + 1 < NCH) store_kv(cur ^ 1);
        __syncthreads();
    }

    // ---- epilogue: normalize, write [B,S,D] f32 ----
#pragma unroll
    for (int r = 0; r < 8; ++r) {
        float inv = 1.0f / lrow[r];
        int s = q0 + wv*16 + r + hi*8;
        float* orow = attn_out + (size_t)(b*Sc + s) * Dc + h*DKc;
#pragma unroll
        for (int tn = 0; tn < 4; ++tn)
            orow[tn*16 + l16] = o[tn][r] * inv;
    }
}

// ---------------------------------------------------------------------------
// Residual + LayerNorm: one block (256 threads) per token row of 1024.
// ---------------------------------------------------------------------------
__global__ __launch_bounds__(256)
void ln_kernel(const float* __restrict__ attn, const float* __restrict__ resid,
               const float* __restrict__ gamma, const float* __restrict__ beta,
               float* __restrict__ out) {
    const int row  = blockIdx.x;
    const int tid  = threadIdx.x;
    const int lane = tid & 31;
    const int wv   = tid >> 5;

    const float* a = attn  + (size_t)row * Dc;
    const float* r = resid + (size_t)row * Dc;

    float y[4], s = 0.0f, ss = 0.0f;
#pragma unroll
    for (int i = 0; i < 4; ++i) {
        int idx = tid + i * 256;
        y[i] = a[idx] + r[idx];
        s  += y[i];
        ss += y[i] * y[i];
    }
#pragma unroll
    for (int off = 16; off > 0; off >>= 1) {
        s  += __shfl_xor(s,  off);
        ss += __shfl_xor(ss, off);
    }
    __shared__ float rs[8], rss[8];
    if (lane == 0) { rs[wv] = s; rss[wv] = ss; }
    __syncthreads();
    float S = 0.0f, SS = 0.0f;
#pragma unroll
    for (int j = 0; j < 8; ++j) { S += rs[j]; SS += rss[j]; }
    float mean = S * (1.0f / Dc);
    float var  = SS * (1.0f / Dc) - mean * mean;
    float rstd = rsqrtf(var + 1e-6f);
#pragma unroll
    for (int i = 0; i < 4; ++i) {
        int idx = tid + i * 256;
        out[(size_t)row * Dc + idx] = (y[i] - mean) * rstd * gamma[idx] + beta[idx];
    }
}

// ---------------------------------------------------------------------------
extern "C" void kernel_launch(void* const* d_in, const int* in_sizes, int n_in,
                              void* d_out, int out_size, void* d_ws, size_t ws_size,
                              hipStream_t stream) {
    (void)in_sizes; (void)n_in; (void)out_size; (void)ws_size;

    const float* query = (const float*)d_in[0];
    const float* key_  = (const float*)d_in[1];
    const float* value = (const float*)d_in[2];
    const float* Wq    = (const float*)d_in[3];
    const float* bq    = (const float*)d_in[4];
    const float* Wk    = (const float*)d_in[5];
    const float* bk    = (const float*)d_in[6];
    const float* Wv    = (const float*)d_in[7];
    const float* bv    = (const float*)d_in[8];
    const float* gamma = (const float*)d_in[9];
    const float* beta  = (const float*)d_in[10];
    float* out = (float*)d_out;

    // workspace: qh/kh/vh (f16, 8MB each) + attn (f32, 16MB) = 40MB
    const size_t nElem = (size_t)Bc * Sc * Dc;   // 4M
    _Float16* qh  = (_Float16*)d_ws;
    _Float16* khp = qh + nElem;
    _Float16* vhp = khp + nElem;
    float* attn = (float*)(vhp + nElem);

    dim3 gGemm(Dc / 128, Mtot / 128);            // 8 x 32
    proj_gemm_kernel<<<gGemm, 256, 0, stream>>>(query, Wq, bq, qh);
    proj_gemm_kernel<<<gGemm, 256, 0, stream>>>(key_,  Wk, bk, khp);
    proj_gemm_kernel<<<gGemm, 256, 0, stream>>>(value, Wv, bv, vhp);

    dim3 gAttn(Sc / 128, Hc, Bc);                // 16 x 16 x 2
    attention_kernel<<<gAttn, 256, 0, stream>>>(qh, khp, vhp, attn);

    ln_kernel<<<Mtot, 256, 0, stream>>>(attn, query, gamma, beta, out);
}
